// Classifier_70772471103881
// MI455X (gfx1250) — compile-verified, matrix operand order checked by reading, not run on hardware
//
#include <hip/hip_runtime.h>
#include <hip/hip_bf16.h>

typedef __attribute__((ext_vector_type(16))) __bf16 v16bf;
typedef __attribute__((ext_vector_type(8)))  float  v8f;
typedef __attribute__((ext_vector_type(4)))  float  v4f;

#define NN   50000
#define EE   800000
#define HH   3
#define DIN  256
#define DHID 64
#define DOUT 10
#define ENC_NEG_INF 0x007FFFFFu   // order-preserving encoding of -inf

__device__ __forceinline__ unsigned enc_f32(float f) {
    unsigned b = __float_as_uint(f);
    return (b & 0x80000000u) ? ~b : (b | 0x80000000u);
}
__device__ __forceinline__ float dec_f32(unsigned u) {
    unsigned b = (u & 0x80000000u) ? (u & 0x7FFFFFFFu) : ~u;
    return __uint_as_float(b);
}

// ---------------------------------------------------------------- init
__global__ void k_init_small(unsigned* m1, float* den1, unsigned* m2, float* den2) {
    int t = blockIdx.x * blockDim.x + threadIdx.x;
    if (t < NN * HH) { m1[t] = ENC_NEG_INF; den1[t] = 0.f; m2[t] = ENC_NEG_INF; den2[t] = 0.f; }
}
__global__ void k_init_zero(float* out1, float* out2) {
    int t = blockIdx.x * blockDim.x + threadIdx.x;
    if (t < NN * DHID) out1[t] = 0.f;
    if (t < NN * DOUT) out2[t] = 0.f;
}

// ---------------------------------------------------------------- GEMM1: z1[N,192] = x[N,256] * W1'[256,192]  (bf16 WMMA, f32 acc)
// Block = 256 threads = 8 waves; wave handles a 16-row tile, 12 col-tiles of 16.
__global__ __launch_bounds__(256) void k_gemm1_wmma(const float* __restrict__ x,
                                                    const float* __restrict__ W1,
                                                    float* __restrict__ z1) {
    __shared__ __align__(32) __bf16 lds_b[12 * 512];   // B fragments for one K-step (12 KB)
    const int tid  = threadIdx.x;
    const int wave = tid >> 5;
    const int lane = tid & 31;
    const int half = lane >> 4;
    const int lr   = lane & 15;
    const int row0 = (blockIdx.x * 8 + wave) * 16;

    v8f c[12];
#pragma unroll
    for (int t = 0; t < 12; ++t)
#pragma unroll
        for (int i = 0; i < 8; ++i) c[t][i] = 0.f;

    int arow = row0 + lr;                 // A layout: lanes 0-15 and 16-31 both cover M=0..15
    if (arow >= NN) arow = NN - 1;        // clamp; OOB rows never stored
    const float* xrow = x + (size_t)arow * DIN;

    for (int ks = 0; ks < 8; ++ks) {      // K = 256 in steps of 32
        __syncthreads();
        // pack W1 (f32) into bf16 B-fragment order for this K-step
        for (int p = tid; p < 12 * 512; p += 256) {
            int t  = p >> 9;
            int r  = p & 511;
            int bl = r >> 4;              // fragment lane
            int j  = r & 15;              // fragment element
            int bh = bl >> 4;
            int nn = bl & 15;
            int col = t * 16 + nn;        // 0..191
            int K  = (j < 8) ? (bh * 8 + j) : (16 + bh * 8 + (j - 8));
            int Kg = ks * 32 + K;
            int h  = col >> 6;
            int e  = col & 63;
            lds_b[p] = (__bf16)W1[((size_t)(h * 256 + Kg)) * 64 + e];
        }
        __syncthreads();

        // A fragment: two contiguous 8-float runs, cvt f32 -> bf16
        int k0 = ks * 32 + half * 8;
        v4f p0 = *(const v4f*)(xrow + k0);
        v4f p1 = *(const v4f*)(xrow + k0 + 4);
        v4f p2 = *(const v4f*)(xrow + k0 + 16);
        v4f p3 = *(const v4f*)(xrow + k0 + 20);
        v16bf a;
#pragma unroll
        for (int j = 0; j < 4; ++j) {
            a[j]      = (__bf16)p0[j];
            a[4 + j]  = (__bf16)p1[j];
            a[8 + j]  = (__bf16)p2[j];
            a[12 + j] = (__bf16)p3[j];
        }
#pragma unroll
        for (int t = 0; t < 12; ++t) {
            v16bf b = *(const v16bf*)(&lds_b[t * 512 + lane * 16]);
            c[t] = __builtin_amdgcn_wmma_f32_16x16x32_bf16(false, a, false, b,
                                                           (short)0, c[t], false, false);
        }
    }

    // store C: VGPR v, lanes 0-15 -> M=v, N=lane ; lanes 16-31 -> M=8+v, N=lane-16
#pragma unroll
    for (int t = 0; t < 12; ++t) {
        int col = t * 16 + lr;
#pragma unroll
        for (int v = 0; v < 8; ++v) {
            int row = row0 + half * 8 + v;
            if (row < NN) z1[(size_t)row * (HH * DHID) + col] = c[t][v];
        }
    }
}

// ---------------------------------------------------------------- per-node attention score dots (layer 1)
__global__ void k_node_scores(const float* __restrict__ z, const float* __restrict__ a_s,
                              const float* __restrict__ a_d, float* __restrict__ es,
                              float* __restrict__ ed) {
    int t = blockIdx.x * blockDim.x + threadIdx.x;
    if (t >= NN * HH) return;
    int n = t / HH, h = t - n * HH;
    const float* zp = z + (size_t)n * (HH * DHID) + h * DHID;
    const float* as = a_s + h * DHID;
    const float* ad = a_d + h * DHID;
    float s = 0.f, d = 0.f;
#pragma unroll 8
    for (int i = 0; i < DHID; ++i) { float v = zp[i]; s += v * as[i]; d += v * ad[i]; }
    es[t] = s; ed[t] = d;
}

// ---------------------------------------------------------------- edge pass A: leaky-relu logit + segment max (generic)
__global__ void k_edge_a(const int* __restrict__ ei, const float* __restrict__ es,
                         const float* __restrict__ ed, unsigned* __restrict__ m,
                         float* __restrict__ ebuf) {
    int e = blockIdx.x * blockDim.x + threadIdx.x;
    if (e >= EE) return;
    int src = ei[e], dst = ei[EE + e];
#pragma unroll
    for (int h = 0; h < HH; ++h) {
        float v = es[src * HH + h] + ed[dst * HH + h];
        v = v > 0.f ? v : 0.2f * v;
        ebuf[e * HH + h] = v;
        atomicMax(&m[dst * HH + h], enc_f32(v));
    }
}

// ---------------------------------------------------------------- edge pass B: exp(e - max) + segment sum (generic)
__global__ void k_edge_b(const int* __restrict__ ei, const unsigned* __restrict__ m,
                         float* __restrict__ ebuf, float* __restrict__ den) {
    int e = blockIdx.x * blockDim.x + threadIdx.x;
    if (e >= EE) return;
    int dst = ei[EE + e];
#pragma unroll
    for (int h = 0; h < HH; ++h) {
        float v = __expf(ebuf[e * HH + h] - dec_f32(m[dst * HH + h]));
        ebuf[e * HH + h] = v;
        atomicAdd(&den[dst * HH + h], v);
    }
}

// ---------------------------------------------------------------- edge pass C (layer 1): 8 threads/edge, 8 dims each
__global__ void k_edge_agg64(const int* __restrict__ ei, const float* __restrict__ ebuf,
                             const float* __restrict__ den, const float* __restrict__ z,
                             float* __restrict__ out) {
    int t = blockIdx.x * blockDim.x + threadIdx.x;
    int e = t >> 3;
    if (e >= EE) return;
    int dl = t & 7;
    int src = ei[e], dst = ei[EE + e];
    float al[HH];
#pragma unroll
    for (int h = 0; h < HH; ++h)
        al[h] = ebuf[e * HH + h] / (den[dst * HH + h] + 1e-16f);
    const float* zp = z + (size_t)src * (HH * DHID);
    float acc[8];
#pragma unroll
    for (int i = 0; i < 8; ++i) acc[i] = 0.f;
#pragma unroll
    for (int h = 0; h < HH; ++h) {
        v4f q0 = *(const v4f*)(zp + h * DHID + dl * 8);
        v4f q1 = *(const v4f*)(zp + h * DHID + dl * 8 + 4);
#pragma unroll
        for (int i = 0; i < 4; ++i) { acc[i] += al[h] * q0[i]; acc[4 + i] += al[h] * q1[i]; }
    }
    float* op = out + (size_t)dst * DHID + dl * 8;
#pragma unroll
    for (int i = 0; i < 8; ++i) atomicAdd(&op[i], acc[i] * (1.f / 3.f));
}

// ---------------------------------------------------------------- ELU + GEMM2 + layer-2 score dots, fused
__global__ __launch_bounds__(256) void k_gemm2_fused(const float* __restrict__ out1,
                                                     const float* __restrict__ W2,
                                                     const float* __restrict__ a_s2,
                                                     const float* __restrict__ a_d2,
                                                     float* __restrict__ z2,
                                                     float* __restrict__ es2,
                                                     float* __restrict__ ed2) {
    __shared__ float w[HH * DHID * DOUT];
    __shared__ float as[HH * DOUT], ad[HH * DOUT];
    int tid = threadIdx.x;
    for (int p = tid; p < HH * DHID * DOUT; p += 256) w[p] = W2[p];
    if (tid < HH * DOUT) { as[tid] = a_s2[tid]; ad[tid] = a_d2[tid]; }
    __syncthreads();
    int n = blockIdx.x * 256 + tid;
    if (n >= NN) return;
    float hv[DHID];
    const float* hp = out1 + (size_t)n * DHID;
#pragma unroll
    for (int i = 0; i < DHID; ++i) { float v = hp[i]; hv[i] = v > 0.f ? v : (__expf(v) - 1.f); }
#pragma unroll
    for (int h = 0; h < HH; ++h) {
        float s = 0.f, d = 0.f;
#pragma unroll
        for (int c = 0; c < DOUT; ++c) {
            const float* wc = &w[(h * DHID) * DOUT + c];
            float acc = 0.f;
#pragma unroll
            for (int i = 0; i < DHID; ++i) acc += hv[i] * wc[i * DOUT];
            z2[(size_t)n * (HH * DOUT) + h * DOUT + c] = acc;
            s += acc * as[h * DOUT + c];
            d += acc * ad[h * DOUT + c];
        }
        es2[n * HH + h] = s; ed2[n * HH + h] = d;
    }
}

// ---------------------------------------------------------------- edge pass C (layer 2): thread/edge, 10 dims
__global__ void k_edge_agg10(const int* __restrict__ ei, const float* __restrict__ ebuf,
                             const float* __restrict__ den, const float* __restrict__ z2,
                             float* __restrict__ out2) {
    int e = blockIdx.x * blockDim.x + threadIdx.x;
    if (e >= EE) return;
    int src = ei[e], dst = ei[EE + e];
    float al[HH];
#pragma unroll
    for (int h = 0; h < HH; ++h)
        al[h] = ebuf[e * HH + h] / (den[dst * HH + h] + 1e-16f);
    const float* zp = z2 + (size_t)src * (HH * DOUT);
    float acc[DOUT];
#pragma unroll
    for (int i = 0; i < DOUT; ++i) acc[i] = 0.f;
#pragma unroll
    for (int h = 0; h < HH; ++h)
#pragma unroll
        for (int i = 0; i < DOUT; ++i) acc[i] += al[h] * zp[h * DOUT + i];
    float* op = out2 + (size_t)dst * DOUT;
#pragma unroll
    for (int i = 0; i < DOUT; ++i) atomicAdd(&op[i], acc[i] * (1.f / 3.f));
}

// ---------------------------------------------------------------- final: h2 and log_softmax(h2)
__global__ void k_finalize(const float* __restrict__ out2, float* __restrict__ dout) {
    int n = blockIdx.x * blockDim.x + threadIdx.x;
    if (n >= NN) return;
    const float* p = out2 + (size_t)n * DOUT;
    float v[DOUT];
    float mx = -3.4e38f;
#pragma unroll
    for (int i = 0; i < DOUT; ++i) { v[i] = p[i]; mx = fmaxf(mx, v[i]); }
    float s = 0.f;
#pragma unroll
    for (int i = 0; i < DOUT; ++i) s += __expf(v[i] - mx);
    float lse = __logf(s) + mx;
#pragma unroll
    for (int i = 0; i < DOUT; ++i) {
        dout[(size_t)n * DOUT + i] = v[i];
        dout[(size_t)NN * DOUT + (size_t)n * DOUT + i] = v[i] - lse;
    }
}

// ---------------------------------------------------------------- launch
extern "C" void kernel_launch(void* const* d_in, const int* in_sizes, int n_in,
                              void* d_out, int out_size, void* d_ws, size_t ws_size,
                              hipStream_t stream) {
    const float* x    = (const float*)d_in[0];
    const int*   ei   = (const int*)d_in[1];
    const float* W1   = (const float*)d_in[2];
    const float* as1  = (const float*)d_in[3];
    const float* ad1  = (const float*)d_in[4];
    const float* W2   = (const float*)d_in[5];
    const float* as2  = (const float*)d_in[6];
    const float* ad2  = (const float*)d_in[7];
    float* dout = (float*)d_out;

    // workspace layout (floats): ~83 MB total, fits in MI455X's 192 MB L2
    float* ws   = (float*)d_ws;
    float*    z1   = ws;                         // N*192
    float*    es1  = z1  + (size_t)NN * HH * DHID;
    float*    ed1  = es1 + NN * HH;
    unsigned* m1   = (unsigned*)(ed1 + NN * HH);
    float*    den1 = (float*)(m1 + NN * HH);
    float*    ex1  = den1 + NN * HH;             // E*3
    float*    out1 = ex1  + (size_t)EE * HH;     // N*64
    float*    z2   = out1 + (size_t)NN * DHID;   // N*30
    float*    es2  = z2   + (size_t)NN * HH * DOUT;
    float*    ed2  = es2  + NN * HH;
    unsigned* m2   = (unsigned*)(ed2 + NN * HH);
    float*    den2 = (float*)(m2 + NN * HH);
    float*    ex2  = den2 + NN * HH;             // E*3
    float*    out2 = ex2  + (size_t)EE * HH;     // N*10

    const int B = 256;
    k_init_small<<<(NN * HH + B - 1) / B, B, 0, stream>>>(m1, den1, m2, den2);
    k_init_zero<<<(NN * DHID + B - 1) / B, B, 0, stream>>>(out1, out2);

    k_gemm1_wmma<<<(NN + 127) / 128, B, 0, stream>>>(x, W1, z1);
    k_node_scores<<<(NN * HH + B - 1) / B, B, 0, stream>>>(z1, as1, ad1, es1, ed1);

    k_edge_a<<<(EE + B - 1) / B, B, 0, stream>>>(ei, es1, ed1, m1, ex1);
    k_edge_b<<<(EE + B - 1) / B, B, 0, stream>>>(ei, m1, ex1, den1);
    k_edge_agg64<<<(EE * 8 + B - 1) / B, B, 0, stream>>>(ei, ex1, den1, z1, out1);

    k_gemm2_fused<<<(NN + B - 1) / B, B, 0, stream>>>(out1, W2, as2, ad2, z2, es2, ed2);

    k_edge_a<<<(EE + B - 1) / B, B, 0, stream>>>(ei, es2, ed2, m2, ex2);
    k_edge_b<<<(EE + B - 1) / B, B, 0, stream>>>(ei, m2, ex2, den2);
    k_edge_agg10<<<(EE + B - 1) / B, B, 0, stream>>>(ei, ex2, den2, z2, out2);

    k_finalize<<<(NN + B - 1) / B, B, 0, stream>>>(out2, dout);
}
